// _LogPolar_Core_76553497083955
// MI455X (gfx1250) — matrix operand order repaired and checked by reading, not compiled, not run on hardware
//
#include <hip/hip_runtime.h>
#include <hip/hip_bf16.h>
#include <math.h>

typedef __attribute__((ext_vector_type(16))) _Float16 v16h;
typedef __attribute__((ext_vector_type(8)))  _Float16 v8h;
typedef __attribute__((ext_vector_type(8)))  float    v8f;
typedef __attribute__((ext_vector_type(4)))  unsigned int u32x4;
typedef __attribute__((ext_vector_type(8)))  int i32x8;
typedef __attribute__((ext_vector_type(4)))  int i32x4;

#define NTAU   64
#define NANG   64
#define IC_    8
#define OC_    32
#define NTAPS  25      // KH*KW
#define KTOT   200     // 25 taps * 8 ic
#define KC_    7       // ceil(200/32) -> K padded to 224 (28 taps)
#define NTILES 240     // 3840 positions / 16
#define SLAB_H 32768   // 64*64*8 halfs per sample (64 KB)
#define AFRAG_H (2*KC_*32*16)  // 7168 halfs
#define TWO_PI_F 6.28318530717958647692f

// ---------------------------------------------------------------------------
// Kernel 0: pack/transpose x (B,IC,T,A,X,Y) f32 -> xt[n][t][a][ic] f16,
// n = b*256 + ij. Reads coalesced along ij; each thread writes one contiguous
// 128B chunk (8 a x 8 ic halfs).
// ---------------------------------------------------------------------------
__global__ __launch_bounds__(256) void pack_x_kernel(
    const float* __restrict__ x, _Float16* __restrict__ xt)
{
  const int bid  = blockIdx.x;        // (b, t, ablk)
  const int ablk = bid & 7;
  const int t    = (bid >> 3) & 63;
  const int b    = bid >> 9;
  const int ij   = threadIdx.x;

  uint4 obuf[8];
  _Float16* bh = reinterpret_cast<_Float16*>(obuf);

  const float* xb = x + (size_t)b * (IC_ * NTAU * NANG * 256)
                      + (size_t)t * (NANG * 256)
                      + (size_t)(ablk * 8) * 256 + ij;
#pragma unroll
  for (int a8 = 0; a8 < 8; ++a8)
#pragma unroll
    for (int ic = 0; ic < 8; ++ic) {
      float v = xb[(size_t)ic * (NTAU * NANG * 256) + a8 * 256];
      bh[a8 * 8 + ic] = (_Float16)v;
    }

  const int nn = b * 256 + ij;
  _Float16* dst = xt + (((size_t)nn * 64 + t) * 64 + ablk * 8) * 8;
  uint4* d4 = reinterpret_cast<uint4*>(dst);
#pragma unroll
  for (int q = 0; q < 8; ++q) d4[q] = obuf[q];
}

// ---------------------------------------------------------------------------
// Kernel 1: weight-norm + pre-swizzle weights into WMMA A-fragment layout.
// K mapping: k = tap*8 + ic, tap = kh*5 + kw (taps 25..27 are zero pad).
// A 16-bit 16x32 fragment: m = lane%16; K(h) = h + 16*(lane/16) (contiguous order, same as B).
// wfrag layout: [oct(2)][kc(7)][lane(32)][h(16)] halfs.
// ---------------------------------------------------------------------------
__global__ __launch_bounds__(256) void prep_weights_kernel(
    const float* __restrict__ conv_v,   // (32, 8, 5, 5) -> oc*200 + ic*25 + tap
    const float* __restrict__ conv_g,   // (32,)
    _Float16* __restrict__ wfrag)
{
  __shared__ float scale[OC_];
  const int tid = threadIdx.x;
  if (tid < OC_) {
    float s = 0.0f;
    for (int k = 0; k < KTOT; ++k) {
      float v = conv_v[tid * KTOT + k];
      s += v * v;
    }
    scale[tid] = conv_g[tid] / sqrtf(s);
  }
  __syncthreads();
  for (int e = tid; e < AFRAG_H; e += blockDim.x) {
    int h    = e & 15;
    int lane = (e >> 4) & 31;
    int kc   = (e >> 9) % KC_;
    int oct  = e / (KC_ * 32 * 16);
    int koff = h + ((lane >> 4) << 4);
    int kg   = kc * 32 + koff;       // 0..223
    int tap  = kg >> 3;
    int ic   = kg & 7;
    int oc   = oct * 16 + (lane & 15);
    float v  = (tap < NTAPS) ? scale[oc] * conv_v[oc * KTOT + ic * NTAPS + tap] : 0.0f;
    wfrag[e] = (_Float16)v;
  }
}

// ---------------------------------------------------------------------------
// Kernel 2: per-sample implicit-GEMM conv via WMMA + fused global max/argmax.
// Slab staged to LDS by the Tensor Data Mover (TENSOR_LOAD_TO_LDS, one 64KB
// 1-D tile, TENSORcnt-tracked); A fragments hoisted to registers (loaded from
// global while the DMA is in flight). B fragment = 2 x ds_load_b128.
// ---------------------------------------------------------------------------
__device__ inline v8h load_tap(const _Float16* slab, int tap, int hrow, int theta) {
  if (tap < NTAPS) {
    int kh  = tap / 5;
    int kw  = tap - kh * 5;
    int off = ((((hrow + kh) << 6) | ((theta + kw) & 63)) << 3); // (tt*64+aa)*8 halfs
    return *reinterpret_cast<const v8h*>(slab + off);
  }
  v8h z = {};
  return z;
}

__global__ __launch_bounds__(256) void conv_wmma_max_kernel(
    const _Float16* __restrict__ xt,    // [1024][64][64][8] f16
    const _Float16* __restrict__ wfrag, // pre-swizzled A fragments (global)
    float* __restrict__ yval,           // [1024][32]
    int*   __restrict__ yidx)           // [1024][32]
{
  extern __shared__ _Float16 dynsmem[];
  _Float16* slab = dynsmem;             // 32768 halfs (64KB)

  const int n    = blockIdx.x;          // sample 0..1023
  const int tid  = threadIdx.x;
  const int wave = tid >> 5;
  const int lane = tid & 31;
  const int col  = lane & 15;
  const int hh   = lane >> 4;           // lane half

#if __has_builtin(__builtin_amdgcn_tensor_load_to_lds)
  // --- TDM path: one descriptor moves the whole 64KB slab global->LDS. ---
  if (wave == 0) {
    const unsigned long long gaddr =
        (unsigned long long)(uintptr_t)(xt + (size_t)n * SLAB_H);
    const unsigned int lds_addr = (unsigned int)(uintptr_t)slab; // low 32 = LDS byte addr
    // D# group0: count=1 (valid), lds_addr, 57-bit global addr, type=2.
    u32x4 g0 = { 1u, lds_addr,
                 (unsigned int)(gaddr & 0xFFFFFFFFu),
                 (unsigned int)((gaddr >> 32) & 0x1FFFFFFu) | 0x80000000u };
    // D# group1: data_size=3 (8B units); tensor_dim0 = tile_dim0 = 8192 units;
    // tensor_dim1 = 1; stride0 = 8192. (1-D contiguous 64KB tile.)
    i32x8 g1 = { 0x00030000,
                 (int)(8192u << 16),   // tensor_dim0[15:0] @ bits 63:48
                 (int)(1u << 16),      // tensor_dim1[15:0] @ bits 95:80
                 (int)(8192u << 16),   // tile_dim0 @ bits 127:112
                 0,                    // tile_dim1/2 unused
                 8192,                 // tensor_dim0_stride[31:0]
                 0, 0 };
    i32x4 gz = { 0, 0, 0, 0 };
#if defined(__clang_major__) && __clang_major__ >= 23
    i32x8 gz8 = { 0, 0, 0, 0, 0, 0, 0, 0 };
    __builtin_amdgcn_tensor_load_to_lds(g0, g1, gz, gz, gz8, 0);
#else
    __builtin_amdgcn_tensor_load_to_lds(g0, g1, gz, gz, 0);
#endif
  }
#else
  // --- Fallback: manual coalesced b128 copy. ---
  {
    const uint4* src = reinterpret_cast<const uint4*>(xt + (size_t)n * SLAB_H);
    uint4* dst = reinterpret_cast<uint4*>(slab);
    for (int i = tid; i < SLAB_H / 8; i += 256) dst[i] = src[i];
  }
#endif

  // Hoist all A fragments into registers (loop-invariant; overlaps the DMA).
  v16h areg[2][KC_];
#pragma unroll
  for (int o = 0; o < 2; ++o)
#pragma unroll
    for (int kc = 0; kc < KC_; ++kc)
      areg[o][kc] = *reinterpret_cast<const v16h*>(
          wfrag + ((o * KC_ + kc) * 32 + lane) * 16);

#if __has_builtin(__builtin_amdgcn_tensor_load_to_lds)
  if (wave == 0) __builtin_amdgcn_s_wait_tensorcnt((short)0);
#endif
  __syncthreads();

  float rmax[2][8];
  int   ridx[2][8];
#pragma unroll
  for (int o = 0; o < 2; ++o)
#pragma unroll
    for (int r = 0; r < 8; ++r) { rmax[o][r] = -__builtin_inff(); ridx[o][r] = 0; }

  for (int t = wave; t < NTILES; t += 8) {
    const int p     = t * 16 + col;     // flat position = hrow*64 + theta
    const int hrow  = p >> 6;
    const int theta = p & 63;
    v8f c0 = {};
    v8f c1 = {};
#pragma unroll
    for (int kc = 0; kc < KC_; ++kc) {
      const int tap0 = (kc << 2) + (hh << 1);
      v8h blo = load_tap(slab, tap0,     hrow, theta);
      v8h bhi = load_tap(slab, tap0 + 1, hrow, theta);
      v16h bfrag = __builtin_shufflevector(blo, bhi,
          0, 1, 2, 3, 4, 5, 6, 7, 8, 9, 10, 11, 12, 13, 14, 15);
      c0 = __builtin_amdgcn_wmma_f32_16x16x32_f16(false, areg[0][kc], false, bfrag,
                                                  (short)0, c0, false, false);
      c1 = __builtin_amdgcn_wmma_f32_16x16x32_f16(false, areg[1][kc], false, bfrag,
                                                  (short)0, c1, false, false);
    }
#pragma unroll
    for (int r = 0; r < 8; ++r) {
      float v0 = c0[r];
      if (v0 > rmax[0][r]) { rmax[0][r] = v0; ridx[0][r] = p; }
      float v1 = c1[r];
      if (v1 > rmax[1][r]) { rmax[1][r] = v1; ridx[1][r] = p; }
    }
  }

  // Reduce across the 16 columns inside each lane-half (wave32 xor shuffles).
#pragma unroll
  for (int o = 0; o < 2; ++o)
#pragma unroll
    for (int r = 0; r < 8; ++r) {
      float v = rmax[o][r];
      int   id = ridx[o][r];
#pragma unroll
      for (int m = 8; m >= 1; m >>= 1) {
        float ov = __shfl_xor(v, m, 32);
        int   oi = __shfl_xor(id, m, 32);
        if (ov > v || (ov == v && oi < id)) { v = ov; id = oi; }
      }
      rmax[o][r] = v;
      ridx[o][r] = id;
    }

  __shared__ float lds_v[OC_][8];
  __shared__ int   lds_i[OC_][8];
  if (col == 0) {
#pragma unroll
    for (int o = 0; o < 2; ++o)
#pragma unroll
      for (int r = 0; r < 8; ++r) {
        int oc = o * 16 + (hh << 3) + r;   // row m = r + 8*half
        lds_v[oc][wave] = rmax[o][r];
        lds_i[oc][wave] = ridx[o][r];
      }
  }
  __syncthreads();
  if (tid < OC_) {
    float v = lds_v[tid][0];
    int   id = lds_i[tid][0];
    for (int w2 = 1; w2 < 8; ++w2) {
      float ov = lds_v[tid][w2];
      int   oi = lds_i[tid][w2];
      if (ov > v || (ov == v && oi < id)) { v = ov; id = oi; }
    }
    yval[n * OC_ + tid] = v;
    yidx[n * OC_ + tid] = id;
  }
}

// ---------------------------------------------------------------------------
// Kernel 3: softmax over the 256 spatial sites + tau / circular-theta encode.
// One block per (b, oc); thread = xy site.
// ---------------------------------------------------------------------------
__global__ __launch_bounds__(256) void finalize_kernel(
    const float* __restrict__ yval,
    const int*   __restrict__ yidx,
    float* __restrict__ out)            // (4, 16, 16, 128)
{
  const int b  = blockIdx.x >> 5;
  const int oc = blockIdx.x & 31;
  const int xy = threadIdx.x;
  const int n  = b * 256 + xy;

  float v  = yval[n * OC_ + oc];
  int   id = yidx[n * OC_ + oc];

  __shared__ float red[256];
  __shared__ float acc[4];

  // stable softmax: max reduce
  red[xy] = v;
  __syncthreads();
  for (int s = 128; s > 0; s >>= 1) {
    if (xy < s) red[xy] = fmaxf(red[xy], red[xy + s]);
    __syncthreads();
  }
  float vmax = red[0];
  __syncthreads();

  float e      = expf(v - vmax);
  float tau_in = (float)id * (1.0f / (64.0f * 64.0f));
  float phase  = TWO_PI_F * (float)(id & 63) * (1.0f / 64.0f);
  float cp     = cosf(phase);
  float sp     = sinf(phase);

  float sums[4] = { e, e * tau_in, e * cp, e * sp };
#pragma unroll
  for (int j = 0; j < 4; ++j) {
    red[xy] = sums[j];
    __syncthreads();
    for (int s = 128; s > 0; s >>= 1) {
      if (xy < s) red[xy] += red[xy + s];
      __syncthreads();
    }
    if (xy == 0) acc[j] = red[0];
    __syncthreads();
  }

  float inv_se = 1.0f / acc[0];
  float mtau   = acc[1] * inv_se;     // sum(weights * tau_in)
  float mcos   = acc[2] * inv_se;
  float msin   = acc[3] * inv_se;
  float rel    = phase - atan2f(msin, mcos);

  size_t base = (size_t)n * 128;
  out[base +       oc] = v;
  out[base +  32 + oc] = tau_in - mtau;
  out[base +  64 + oc] = cosf(rel);
  out[base +  96 + oc] = sinf(rel);
}

// ---------------------------------------------------------------------------
extern "C" void kernel_launch(void* const* d_in, const int* in_sizes, int n_in,
                              void* d_out, int out_size, void* d_ws, size_t ws_size,
                              hipStream_t stream) {
  const float* x      = (const float*)d_in[0];   // (4,8,64,64,16,16) f32
  const float* conv_v = (const float*)d_in[1];   // (32,8,5,5) f32
  const float* conv_g = (const float*)d_in[2];   // (32,) f32
  float* out = (float*)d_out;                    // 4*16*16*128 f32

  char* ws = (char*)d_ws;
  _Float16* wfrag = (_Float16*)ws;                         // 14336 B (pad to 16384)
  float*    yval  = (float*)(ws + 16384);                  // 131072 B
  int*      yidx  = (int*)(ws + 16384 + 131072);           // 131072 B
  _Float16* xt    = (_Float16*)(ws + 16384 + 2 * 131072);  // 64 MiB packed f16

  pack_x_kernel<<<4 * 64 * 8, 256, 0, stream>>>(x, xt);
  prep_weights_kernel<<<1, 256, 0, stream>>>(conv_v, conv_g, wfrag);

  const size_t dyn_lds = (size_t)SLAB_H * sizeof(_Float16); // 65536 B
  conv_wmma_max_kernel<<<1024, 256, dyn_lds, stream>>>(xt, wfrag, yval, yidx);

  finalize_kernel<<<4 * OC_, 256, 0, stream>>>(yval, yidx, out);
}
